// SoftSOMLayer_67894843015429
// MI455X (gfx1250) — matrix operand order repaired and checked by reading, not compiled, not run on hardware
//
#include <hip/hip_runtime.h>
#include <hip/hip_bf16.h>
#include <stdint.h>

#define BATCH  16384
#define DIM    1024
#define NUNITS 1024
#define GDIM   32
#define TILE_M 32          // 32 batch rows per block: each B fragment feeds 2 WMMAs

typedef __attribute__((ext_vector_type(16))) __bf16 v16bf;
typedef __attribute__((ext_vector_type(8)))  __bf16 v8bf;
typedef __attribute__((ext_vector_type(8)))  float  v8f;
typedef __attribute__((ext_vector_type(4)))  unsigned short us4;

__device__ __forceinline__ unsigned short f2bf_rne(float f) {
  unsigned int u = __float_as_uint(f);
  u += 0x7FFFu + ((u >> 16) & 1u);   // round to nearest even
  return (unsigned short)(u >> 16);
}
__device__ __forceinline__ __bf16 bits2bf(unsigned short s) {
  union { unsigned short u; __bf16 b; } c; c.u = s; return c.b;
}

// B fragment: 16 contiguous bf16 (lane holds column = lane%16, K contiguous).
__device__ __forceinline__ v16bf load_bf16x16(const __bf16* p) {
  union { v16bf v; v8bf h[2]; } u;
  u.h[0] = *(const v8bf*)(p);
  u.h[1] = *(const v8bf*)(p + 8);
  return u.v;
}
// A fragment: 16x32 bf16 tile, lane = row (mod 16); contiguous K to match B:
// lane<16 holds K {0..15}, lane>=16 holds K {16..31}.
__device__ __forceinline__ v16bf load_a_frag(const __bf16* row, int kb, int hi16) {
  union { v16bf v; v8bf h[2]; } u;
  const __bf16* p = row + kb + 16 * hi16;
  u.h[0] = *(const v8bf*)(p);
  u.h[1] = *(const v8bf*)(p + 8);
  return u.v;
}

// ---------------- prep: x -> bf16 ----------------
__global__ __launch_bounds__(256) void som_prep_x(const float* __restrict__ x,
                                                  unsigned short* __restrict__ xb,
                                                  int n4) {
  int i = blockIdx.x * 256 + threadIdx.x;
  if (i < n4) {
    float4 f = ((const float4*)x)[i];
    us4 o;
    o[0] = f2bf_rne(f.x); o[1] = f2bf_rne(f.y);
    o[2] = f2bf_rne(f.z); o[3] = f2bf_rne(f.w);
    ((us4*)xb)[i] = o;
  }
}

// ---------------- prep: w -> bf16, wT -> bf16, ||w||^2 ----------------
__global__ __launch_bounds__(256) void som_prep_w(const float* __restrict__ w,
                                                  unsigned short* __restrict__ wb,
                                                  unsigned short* __restrict__ wtb,
                                                  float* __restrict__ wsq) {
  const int n = blockIdx.x;          // unit index
  const int tid = threadIdx.x;
  const float* row = w + (size_t)n * DIM;
  float s = 0.f;
  for (int d = tid; d < DIM; d += 256) {
    float f = row[d];
    s += f * f;
    unsigned short b = f2bf_rne(f);
    wb[(size_t)n * DIM + d] = b;          // [unit][dim]  (B for GEMM1)
    wtb[(size_t)d * NUNITS + n] = b;      // [dim][unit]  (B for GEMM2)
  }
  __shared__ float red[256];
  red[tid] = s;
  __syncthreads();
  for (int off = 128; off > 0; off >>= 1) {
    if (tid < off) red[tid] += red[tid + off];
    __syncthreads();
  }
  if (tid == 0) wsq[n] = red[0];
}

// ---------------- fused: GEMM1 + dual softmax + GEMM2 ----------------
__global__ __launch_bounds__(256) void som_fused(const __bf16* __restrict__ xb,
                                                 const __bf16* __restrict__ wb,
                                                 const __bf16* __restrict__ wtb,
                                                 const float*  __restrict__ wsq,
                                                 float* __restrict__ out) {
  __shared__ float  s_logits[TILE_M][NUNITS];      // 128 KB (reused for combined f32)
  __shared__ __bf16 s_comb[TILE_M][NUNITS];        // 64 KB
  __shared__ float  s_rmax[TILE_M][GDIM], s_rsum[TILE_M][GDIM]; // axis2 (g2 contiguous)
  __shared__ float  s_cmax[TILE_M][GDIM], s_csum[TILE_M][GDIM]; // axis1 (g1 strided)
  __shared__ float  s_part[TILE_M][8];
  __shared__ float  s_inv[TILE_M];

  const int tid  = threadIdx.x;
  const int wave = tid >> 5;           // 0..7
  const int lane = tid & 31;
  const int hi16 = lane >> 4;
  const int ln   = lane & 15;
  const int row0 = blockIdx.x * TILE_M;

  const __bf16* xrow0 = xb + (size_t)(row0 + ln) * DIM;        // A half 0
  const __bf16* xrow1 = xb + (size_t)(row0 + 16 + ln) * DIM;   // A half 1

  // ===== GEMM1: dot[b,n] = x_b . w_n  (32 rows x 1024 units, K = 1024) =====
  v8f acc[2][8] = {};
  const int ncol0 = wave * 128;
  for (int kb = 0; kb < DIM; kb += 32) {
    __builtin_prefetch(xrow0 + kb + 64, 0, 3);           // global_prefetch_b8
    __builtin_prefetch(xrow1 + kb + 64, 0, 3);
    v16bf a0 = load_a_frag(xrow0, kb, hi16);
    v16bf a1 = load_a_frag(xrow1, kb, hi16);
#pragma unroll
    for (int t = 0; t < 8; ++t) {
      const __bf16* bp = wb + (size_t)(ncol0 + t * 16 + ln) * DIM + kb + 16 * hi16;
      v16bf b = load_bf16x16(bp);      // one B fragment feeds two WMMAs
      acc[0][t] = __builtin_amdgcn_wmma_f32_16x16x32_bf16(
          false, a0, false, b, (short)0, acc[0][t], false, false);
      acc[1][t] = __builtin_amdgcn_wmma_f32_16x16x32_bf16(
          false, a1, false, b, (short)0, acc[1][t], false, false);
    }
  }

  // effective logits: 2*dot - ||w||^2  (||x||^2 cancels in all softmaxes)
#pragma unroll
  for (int t = 0; t < 8; ++t) {
    const int n = ncol0 + t * 16 + ln;
    const float wq = wsq[n];
#pragma unroll
    for (int r = 0; r < 8; ++r) {
      s_logits[r + 8 * hi16][n]      = 2.0f * acc[0][t][r] - wq;
      s_logits[16 + r + 8 * hi16][n] = 2.0f * acc[1][t][r] - wq;
    }
  }
  __syncthreads();

  // ===== softmax stats: axis2 (contiguous g2) and axis1 (stride-32 g1) =====
  for (int task = tid; task < TILE_M * GDIM; task += 256) {
    const int m  = task >> 5;
    const int g1 = task & 31;
    const float* p = &s_logits[m][g1 * GDIM];
    // lane-rotated reads: conflict-free across the wave
    float mx = -3.4e38f;
    for (int i = 0; i < GDIM; ++i) mx = fmaxf(mx, p[(i + lane) & 31]);
    float sm = 0.f;
    for (int i = 0; i < GDIM; ++i) sm += __expf(p[(i + lane) & 31] - mx);
    s_rmax[m][g1] = mx; s_rsum[m][g1] = sm;
  }
  for (int task = tid; task < TILE_M * GDIM; task += 256) {
    const int m  = task >> 5;
    const int g2 = task & 31;
    const float* p = &s_logits[m][g2];
    float mx = -3.4e38f;
    for (int i = 0; i < GDIM; ++i) mx = fmaxf(mx, p[i * GDIM]);
    float sm = 0.f;
    for (int i = 0; i < GDIM; ++i) sm += __expf(p[i * GDIM] - mx);
    s_cmax[m][g2] = mx; s_csum[m][g2] = sm;
  }
  __syncthreads();

  // ===== combine s1*s2, per-batch-row total (8 threads per row) =====
  {
    const int m   = tid >> 3;        // batch row in tile (0..31)
    const int seg = tid & 7;         // 128 contiguous elements each
    float part = 0.f;
    for (int i = 0; i < 128; ++i) {
      const int n  = seg * 128 + ((i + 4 * lane) & 127);
      const int g1 = n >> 5, g2 = n & 31;
      const float l = s_logits[m][n];
      const float v = (__expf(l - s_cmax[m][g2]) / s_csum[m][g2]) *
                      (__expf(l - s_rmax[m][g1]) / s_rsum[m][g1]);
      s_logits[m][n] = v;
      part += v;
    }
    s_part[m][seg] = part;
  }
  __syncthreads();
  if (tid < TILE_M) {
    float tot = 0.f;
#pragma unroll
    for (int i = 0; i < 8; ++i) tot += s_part[tid][i];
    s_inv[tid] = 1.0f / (tot + 1e-8f);
  }
  __syncthreads();
  {
    const int m   = tid >> 3;
    const int seg = tid & 7;
    const float inv = s_inv[m];
    for (int i = 0; i < 128; ++i) {
      const int n = seg * 128 + ((i + 4 * lane) & 127);
      s_comb[m][n] = bits2bf(f2bf_rne(s_logits[m][n] * inv));
    }
  }
  __syncthreads();

  // ===== GEMM2: recon[b,d] = combined[b,:] @ wT[:,d]  (A from LDS) =====
  v8f acc2[2][8] = {};
  const __bf16* crow0 = &s_comb[ln][0];
  const __bf16* crow1 = &s_comb[16 + ln][0];
  const int dcol0 = wave * 128;
  for (int kb = 0; kb < NUNITS; kb += 32) {
    v16bf a0 = load_a_frag(crow0, kb, hi16);             // ds_load_b128 pairs
    v16bf a1 = load_a_frag(crow1, kb, hi16);
#pragma unroll
    for (int t = 0; t < 8; ++t) {
      const __bf16* bp = wtb + (size_t)(dcol0 + t * 16 + ln) * NUNITS + kb + 16 * hi16;
      v16bf b = load_bf16x16(bp);
      acc2[0][t] = __builtin_amdgcn_wmma_f32_16x16x32_bf16(
          false, a0, false, b, (short)0, acc2[0][t], false, false);
      acc2[1][t] = __builtin_amdgcn_wmma_f32_16x16x32_bf16(
          false, a1, false, b, (short)0, acc2[1][t], false, false);
    }
  }
#pragma unroll
  for (int t = 0; t < 8; ++t) {
    const int d = dcol0 + t * 16 + ln;
#pragma unroll
    for (int r = 0; r < 8; ++r) {
      out[(size_t)(row0 + r + 8 * hi16) * DIM + d]      = acc2[0][t][r];
      out[(size_t)(row0 + 16 + r + 8 * hi16) * DIM + d] = acc2[1][t][r];
    }
  }
}

extern "C" void kernel_launch(void* const* d_in, const int* in_sizes, int n_in,
                              void* d_out, int out_size, void* d_ws, size_t ws_size,
                              hipStream_t stream) {
  const float* x = (const float*)d_in[0];        // [16384, 1024] f32
  const float* w = (const float*)d_in[1];        // [32, 32, 1024] f32
  float* out = (float*)d_out;                    // [16384, 1024] f32

  char* ws = (char*)d_ws;
  const size_t XB_BYTES  = (size_t)BATCH * DIM * 2;      // 32 MB
  const size_t WB_BYTES  = (size_t)NUNITS * DIM * 2;     // 2 MB
  unsigned short* xb  = (unsigned short*)(ws);
  unsigned short* wb  = (unsigned short*)(ws + XB_BYTES);
  unsigned short* wtb = (unsigned short*)(ws + XB_BYTES + WB_BYTES);
  float*          wsq = (float*)(ws + XB_BYTES + 2 * WB_BYTES);

  const int n4 = BATCH * DIM / 4;
  som_prep_x<<<n4 / 256, 256, 0, stream>>>(x, xb, n4);
  som_prep_w<<<NUNITS, 256, 0, stream>>>(w, wb, wtb, wsq);
  som_fused<<<BATCH / TILE_M, 256, 0, stream>>>(
      (const __bf16*)xb, (const __bf16*)wb, (const __bf16*)wtb, wsq, out);
}
